// SoftMoE9_3719441678656
// MI455X (gfx1250) — compile-verified, matrix operand order checked by reading, not run on hardware
//
#include <hip/hip_runtime.h>
#include <math.h>

#define DEVI __device__ __forceinline__

typedef __bf16 bf16;
typedef __attribute__((ext_vector_type(16))) __bf16 v16bf;
typedef __attribute__((ext_vector_type(2)))  __bf16 bf16x2;
typedef __attribute__((ext_vector_type(8)))  float  v8f;

// Problem dims (fixed by reference)
constexpr int BB = 4;
constexpr int NN = 4096;
constexpr int DD = 1024;
constexpr int EE = 8;
constexpr int SS = 512;
constexpr int ESS = EE * SS;     // 4096
constexpr int HH = 4 * DD / EE;  // 512
constexpr float FEPS = 1e-12f;

constexpr int LDA = 40;  // padded A-tile row stride (80B: bank-spread, 16B aligned)

#define WMMA_BF16(a, b, c) \
  __builtin_amdgcn_wmma_f32_16x16x32_bf16(false, (a), false, (b), (short)0, (c), false, false)

DEVI v8f vzero8() { v8f z = {0.f,0.f,0.f,0.f,0.f,0.f,0.f,0.f}; return z; }

// bf16 -> f32 expansion without v_cvt: bf16 is the high 16 bits of f32.
DEVI float bflo(unsigned u) { union { unsigned x; float f; } c; c.x = u << 16; return c.f; }
DEVI float bfhi(unsigned u) { union { unsigned x; float f; } c; c.x = u & 0xffff0000u; return c.f; }

// ---- CDNA5 async global->LDS copy (ASYNCcnt-tracked, no VGPR data) --------
DEVI void async_ld_b128(void* lds, const void* g) {
  asm volatile("global_load_async_to_lds_b128 %0, %1, off"
               :: "v"((unsigned int)(unsigned long long)(uintptr_t)lds), "v"(g)
               : "memory");
}
DEVI void async_wait0() { asm volatile("s_wait_asynccnt 0" ::: "memory"); }

// ---- WMMA fragment handling (wave32 layouts, ISA 7.12.2) ------------------
// A: 16x32 bf16 row-major in LDS (padded rows). lane L: M = L%16, half = L/16.
// Per lane this is two contiguous 16B runs -> 2x ds_load_b128.
DEVI v16bf frag_a(const bf16* A, int lda, int m0, int lane) {
  v16bf f;
  const bf16* row = A + (size_t)(m0 + (lane & 15)) * lda;
  int half8 = (lane >> 4) * 8;
#pragma unroll
  for (int v = 0; v < 8; ++v) {
    int k = ((v < 4) ? (2 * v) : (16 + 2 * (v - 4))) + half8;
    f[2 * v]     = row[k];
    f[2 * v + 1] = row[k + 1];
  }
  return f;
}

// B tile stored SWIZZLED in fragment order: element (k,n), k in [0,32), n in
// [0,64) lives at  [( (n>>4)*32 + (n&15) + 16*((k>>3)&1) )*16 + elem(k)].
// Each lane's fragment is then 32 contiguous bytes -> 2x ds_load_b128.
DEVI int bswz(int k, int n) {
  int half = (k >> 3) & 1;
  int kb = k - 8 * half;                               // {0..7} U {16..23}
  int v = (kb < 8) ? (kb >> 1) : (4 + ((kb - 16) >> 1));
  int elem = 2 * v + (kb & 1);
  return (((n >> 4) * 32 + (n & 15) + 16 * half) << 4) + elem;
}
// For even k, (k,n) and (k+1,n) are consecutive elements -> one 4B store.
DEVI void store_bpair(bf16* Bt, int keven, int n, bf16 lo, bf16 hi) {
  bf16x2 v = {lo, hi};
  *(bf16x2*)(Bt + bswz(keven, n)) = v;
}
DEVI v16bf frag_b_swz(const bf16* Bt, int j, int lane) {
  return *(const v16bf*)(Bt + (((j * 32 + lane)) << 4));
}

// One K=32 step: block tile 128x64, 4 waves; wave wm covers rows wm*32..+31,
// all 64 cols -> 2x4 C fragments, 8 WMMAs.
DEVI void mma_128x64(const bf16* A, const bf16* Bt, int lane, int wm,
                     v8f acc[2][4]) {
  v16bf a0 = frag_a(A, LDA, wm * 32, lane);
  v16bf a1 = frag_a(A, LDA, wm * 32 + 16, lane);
  v16bf bf[4];
#pragma unroll
  for (int j = 0; j < 4; ++j) bf[j] = frag_b_swz(Bt, j, lane);
#pragma unroll
  for (int j = 0; j < 4; ++j) {
    acc[0][j] = WMMA_BF16(a0, bf[j], acc[0][j]);
    acc[1][j] = WMMA_BF16(a1, bf[j], acc[1][j]);
  }
}

// C/D 16x16 f32 layout: lane L: N = L%16; VGPR r: M = r + 8*(L/16)
DEVI void store_c_bf(bf16* dst, int ld, int m0, int n0, const v8f c, int lane) {
  int n = n0 + (lane & 15);
  int mh = m0 + ((lane >> 4) ? 8 : 0);
#pragma unroll
  for (int r = 0; r < 8; ++r) dst[(size_t)(mh + r) * ld + n] = (bf16)c[r];
}
DEVI void store_c_f32(float* dst, int ld, int m0, int n0, const v8f c, int lane) {
  int n = n0 + (lane & 15);
  int mh = m0 + ((lane >> 4) ? 8 : 0);
#pragma unroll
  for (int r = 0; r < 8; ++r) dst[(size_t)(mh + r) * ld + n] = c[r];
}

// ===========================================================================
// K0a: rnx = 1 / max(||x row||, eps)   (one wave32 per row)
__global__ __launch_bounds__(256) void k_rnx(const float* __restrict__ x,
                                             float* __restrict__ rnx) {
  int wave = threadIdx.x >> 5, lane = threadIdx.x & 31;
  int row = blockIdx.x * 8 + wave;
  const float* xr = x + (size_t)row * DD;
  float s = 0.f;
  for (int i = lane; i < DD; i += 32) {
    float v = xr[i];
    s += v * v;
  }
#pragma unroll
  for (int m = 16; m; m >>= 1) s += __shfl_xor(s, m);
  if (lane == 0) rnx[row] = 1.0f / fmaxf(sqrtf(s), FEPS);
}

// K0b: cmu[es] = scale / max(||mu col||, eps)
__global__ __launch_bounds__(256) void k_cmu(const float* __restrict__ mu,
                                             const float* __restrict__ scale,
                                             float* __restrict__ cmu) {
  int es = blockIdx.x * 256 + threadIdx.x;
  float s = 0.f;
  for (int d = 0; d < DD; ++d) {
    float v = mu[(size_t)d * ESS + es];
    s += v * v;
  }
  cmu[es] = scale[0] / fmaxf(sqrtf(s), FEPS);
}

// K0c: fp32 -> bf16 elementwise (w1, w2)
__global__ __launch_bounds__(256) void k_cvt_bf16(const float* __restrict__ in,
                                                  bf16* __restrict__ out, int n) {
  int i = blockIdx.x * 256 + threadIdx.x;
  if (i < n) out[i] = (bf16)in[i];
}

// ===========================================================================
// GEMM1: logits = x_norm @ mu_norm, bf16 out.  K = D.
__global__ __launch_bounds__(128) void k_gemm_logits(
    const float* __restrict__ x, const float* __restrict__ mu,
    const float* __restrict__ rnx, const float* __restrict__ cmu,
    bf16* __restrict__ logits) {
  __shared__ bf16 As[2][128 * LDA];
  __shared__ bf16 Bs[2][32 * 64];
  int tid = threadIdx.x, lane = tid & 31, wm = tid >> 5;
  int b = blockIdx.z;
  int m0 = blockIdx.y * 128;  // token tile
  int n0 = blockIdx.x * 64;   // es tile
  v8f acc[2][4];
#pragma unroll
  for (int i = 0; i < 2; ++i)
#pragma unroll
    for (int j = 0; j < 4; ++j) acc[i][j] = vzero8();

  int kp = tid >> 3;         // 16 k-pairs (k = 2kp, 2kp+1)
  int ng = (tid & 7) * 8;    // 8 consecutive n
  const float ascale = rnx[b * NN + m0 + tid];
  float cs8[8];
#pragma unroll
  for (int i = 0; i < 8; ++i) cs8[i] = cmu[n0 + ng + i];
  const float* xbase = x + ((size_t)(b * NN + m0 + tid)) * DD;

  auto stage = [&](int buf, int k0) {
    const float4* xs = (const float4*)(xbase + k0);
    bf16* ar = &As[buf][tid * LDA];
#pragma unroll
    for (int i = 0; i < 8; ++i) {
      float4 v = xs[i];
      ar[i * 4 + 0] = (bf16)(v.x * ascale);
      ar[i * 4 + 1] = (bf16)(v.y * ascale);
      ar[i * 4 + 2] = (bf16)(v.z * ascale);
      ar[i * 4 + 3] = (bf16)(v.w * ascale);
    }
    const float* r0p = mu + (size_t)(k0 + 2 * kp) * ESS + n0 + ng;
    const float* r1p = r0p + ESS;
    float r0[8], r1[8];
    *(float4*)&r0[0] = ((const float4*)r0p)[0];
    *(float4*)&r0[4] = ((const float4*)r0p)[1];
    *(float4*)&r1[0] = ((const float4*)r1p)[0];
    *(float4*)&r1[4] = ((const float4*)r1p)[1];
    bf16* bt = &Bs[buf][0];
#pragma unroll
    for (int i = 0; i < 8; ++i)
      store_bpair(bt, 2 * kp, ng + i, (bf16)(r0[i] * cs8[i]),
                  (bf16)(r1[i] * cs8[i]));
  };

  stage(0, 0);
  __syncthreads();
  constexpr int NK = DD / 32;
  for (int it = 0; it < NK; ++it) {
    int cur = it & 1;
    if (it + 1 < NK) stage(cur ^ 1, (it + 1) * 32);
    mma_128x64(As[cur], Bs[cur], lane, wm, acc);
    __syncthreads();
  }
  bf16* outb = logits + (size_t)b * NN * ESS;
#pragma unroll
  for (int i = 0; i < 2; ++i)
#pragma unroll
    for (int j = 0; j < 4; ++j)
      store_c_bf(outb, ESS, m0 + wm * 32 + i * 16, n0 + j * 16, acc[i][j], lane);
}

// ===========================================================================
// Row stats (combine softmax, over ES): one wave per (b,n) row.
__global__ __launch_bounds__(256) void k_rowstats(const bf16* __restrict__ logits,
                                                  float* __restrict__ rowmax,
                                                  float* __restrict__ rowinv) {
  int wave = threadIdx.x >> 5, lane = threadIdx.x & 31;
  int row = blockIdx.x * 8 + wave;
  const bf16* lr = logits + (size_t)row * ESS;
  float m = -1e30f;
  for (int i = lane; i < ESS; i += 32) m = fmaxf(m, (float)lr[i]);
#pragma unroll
  for (int k = 16; k; k >>= 1) m = fmaxf(m, __shfl_xor(m, k));
  float s = 0.f;
  for (int i = lane; i < ESS; i += 32) s += __expf((float)lr[i] - m);
#pragma unroll
  for (int k = 16; k; k >>= 1) s += __shfl_xor(s, k);
  if (lane == 0) {
    rowmax[row] = m;
    rowinv[row] = 1.0f / s;
  }
}

// Column stats (dispatch softmax, over N): coalesced column walk.
__global__ __launch_bounds__(256) void k_colstats(const bf16* __restrict__ logits,
                                                  float* __restrict__ colmax,
                                                  float* __restrict__ colinv) {
  int es = blockIdx.x * 256 + threadIdx.x;
  int b = blockIdx.y;
  const bf16* base = logits + (size_t)b * NN * ESS + es;
  float m = -1e30f;
  for (int n = 0; n < NN; ++n) m = fmaxf(m, (float)base[(size_t)n * ESS]);
  float s = 0.f;
  for (int n = 0; n < NN; ++n) s += __expf((float)base[(size_t)n * ESS] - m);
  colmax[b * ESS + es] = m;
  colinv[b * ESS + es] = 1.0f / s;
}

// ===========================================================================
// GEMM2: slot_in = dispatch^T @ x  (ES x D, K = N).
__global__ __launch_bounds__(128) void k_gemm_slot_in(
    const bf16* __restrict__ logits, const float* __restrict__ x,
    const float* __restrict__ colmax, const float* __restrict__ colinv,
    bf16* __restrict__ slot_in) {
  __shared__ bf16 As[2][128 * LDA];
  __shared__ bf16 Bs[2][32 * 64];
  int tid = threadIdx.x, lane = tid & 31, wm = tid >> 5;
  int b = blockIdx.z;
  int m0 = blockIdx.y * 128;  // es tile (M)
  int n0 = blockIdx.x * 64;   // d tile (N)
  v8f acc[2][4];
#pragma unroll
  for (int i = 0; i < 2; ++i)
#pragma unroll
    for (int j = 0; j < 4; ++j) acc[i][j] = vzero8();

  int seg = tid & 7, nl = tid >> 3;  // A: 8 es-groups of 16 x 16 n-rows
  int kp = tid >> 3, ng = (tid & 7) * 8;  // B pairs
  float cm[16], ci[16];
#pragma unroll
  for (int i = 0; i < 16; ++i) {
    int es = m0 + seg * 16 + i;
    cm[i] = colmax[b * ESS + es];
    ci[i] = colinv[b * ESS + es];
  }

  auto stage = [&](int buf, int k0) {
#pragma unroll
    for (int j = 0; j < 2; ++j) {
      int n = nl + j * 16;
      const uint4* lp = (const uint4*)(logits +
          ((size_t)(b * NN + k0 + n)) * ESS + m0 + seg * 16);
      uint4 u0 = lp[0], u1 = lp[1];
      unsigned uu[8] = {u0.x, u0.y, u0.z, u0.w, u1.x, u1.y, u1.z, u1.w};
#pragma unroll
      for (int t = 0; t < 8; ++t) {
        float plo = __expf(bflo(uu[t]) - cm[2 * t]) * ci[2 * t];
        float phi = __expf(bfhi(uu[t]) - cm[2 * t + 1]) * ci[2 * t + 1];
        As[buf][(seg * 16 + 2 * t) * LDA + n] = (bf16)plo;       // [es][n]
        As[buf][(seg * 16 + 2 * t + 1) * LDA + n] = (bf16)phi;
      }
    }
    const float* r0p = x + ((size_t)(b * NN + k0 + 2 * kp)) * DD + n0 + ng;
    const float* r1p = r0p + DD;
    float r0[8], r1[8];
    *(float4*)&r0[0] = ((const float4*)r0p)[0];
    *(float4*)&r0[4] = ((const float4*)r0p)[1];
    *(float4*)&r1[0] = ((const float4*)r1p)[0];
    *(float4*)&r1[4] = ((const float4*)r1p)[1];
    bf16* bt = &Bs[buf][0];
#pragma unroll
    for (int i = 0; i < 8; ++i)
      store_bpair(bt, 2 * kp, ng + i, (bf16)r0[i], (bf16)r1[i]);
  };

  stage(0, 0);
  __syncthreads();
  constexpr int NK = NN / 32;
  for (int it = 0; it < NK; ++it) {
    int cur = it & 1;
    if (it + 1 < NK) stage(cur ^ 1, (it + 1) * 32);
    mma_128x64(As[cur], Bs[cur], lane, wm, acc);
    __syncthreads();
  }
  bf16* outb = slot_in + (size_t)b * ESS * DD;
#pragma unroll
  for (int i = 0; i < 2; ++i)
#pragma unroll
    for (int j = 0; j < 4; ++j)
      store_c_bf(outb, DD, m0 + wm * 32 + i * 16, n0 + j * 16, acc[i][j], lane);
}

// ===========================================================================
// GEMM3a: h = gelu(slot_in @ w1[e]^T + b1[e]).  K = D.  A-tile via async DMA.
__global__ __launch_bounds__(128) void k_gemm_mlp1(
    const bf16* __restrict__ slot_in, const bf16* __restrict__ w1b,
    const float* __restrict__ b1, bf16* __restrict__ hbuf) {
  __shared__ bf16 As[2][128 * LDA];
  __shared__ bf16 Bs[2][32 * 64];
  int tid = threadIdx.x, lane = tid & 31, wm = tid >> 5;
  int b = blockIdx.z;
  int m0 = blockIdx.y * 128;  // es tile
  int n0 = blockIdx.x * 64;   // h tile
  int e = m0 / SS;
  v8f acc[2][4];
#pragma unroll
  for (int i = 0; i < 2; ++i)
#pragma unroll
    for (int j = 0; j < 4; ++j) acc[i][j] = vzero8();

  int hl = tid >> 1, hseg = tid & 1;  // B: 64 h-cols x 2 k-segs of 16
  const bf16* abase = slot_in + ((size_t)(b * ESS + m0 + tid)) * DD;
  const bf16* wbase = w1b + ((size_t)(e * HH + n0 + hl)) * DD + hseg * 16;

  auto stage = [&](int buf, int k0) {
    bf16* ar = &As[buf][tid * LDA];
    const bf16* as = abase + k0;  // 64B contiguous -> 4x b128 async DMA
#pragma unroll
    for (int i = 0; i < 4; ++i) async_ld_b128(ar + i * 8, as + i * 8);
    const bf16* wr = wbase + k0;  // 16 consecutive k for fixed n=hl
    bf16* bt = &Bs[buf][0];
#pragma unroll
    for (int i = 0; i < 8; ++i)
      store_bpair(bt, hseg * 16 + 2 * i, hl, wr[2 * i], wr[2 * i + 1]);
  };

  stage(0, 0);
  async_wait0();
  __syncthreads();
  constexpr int NK = DD / 32;
  for (int it = 0; it < NK; ++it) {
    int cur = it & 1;
    if (it + 1 < NK) stage(cur ^ 1, (it + 1) * 32);
    mma_128x64(As[cur], Bs[cur], lane, wm, acc);
    async_wait0();
    __syncthreads();
  }
  bf16* outb = hbuf + (size_t)b * ESS * HH;
#pragma unroll
  for (int i = 0; i < 2; ++i)
#pragma unroll
    for (int j = 0; j < 4; ++j) {
      int nn = n0 + j * 16 + (lane & 15);
      int mh = m0 + wm * 32 + i * 16 + ((lane >> 4) ? 8 : 0);
      float bias = b1[e * HH + nn];
#pragma unroll
      for (int r = 0; r < 8; ++r) {
        float v = acc[i][j][r] + bias;
        float g = 0.5f * v * (1.0f + erff(v * 0.70710678118f));  // exact gelu
        outb[(size_t)(mh + r) * HH + nn] = (bf16)g;
      }
    }
}

// GEMM3b: slot_out = h @ w2[e]^T + b2[e].  K = H.  A-tile via async DMA.
__global__ __launch_bounds__(128) void k_gemm_mlp2(
    const bf16* __restrict__ hbuf, const bf16* __restrict__ w2b,
    const float* __restrict__ b2, bf16* __restrict__ slot_out) {
  __shared__ bf16 As[2][128 * LDA];
  __shared__ bf16 Bs[2][32 * 64];
  int tid = threadIdx.x, lane = tid & 31, wm = tid >> 5;
  int b = blockIdx.z;
  int m0 = blockIdx.y * 128;  // es tile
  int n0 = blockIdx.x * 64;   // d tile
  int e = m0 / SS;
  v8f acc[2][4];
#pragma unroll
  for (int i = 0; i < 2; ++i)
#pragma unroll
    for (int j = 0; j < 4; ++j) acc[i][j] = vzero8();

  int dl = tid >> 1, dseg = tid & 1;
  const bf16* abase = hbuf + ((size_t)(b * ESS + m0 + tid)) * HH;
  const bf16* wbase = w2b + ((size_t)(e * DD + n0 + dl)) * HH + dseg * 16;

  auto stage = [&](int buf, int k0) {
    bf16* ar = &As[buf][tid * LDA];
    const bf16* as = abase + k0;
#pragma unroll
    for (int i = 0; i < 4; ++i) async_ld_b128(ar + i * 8, as + i * 8);
    const bf16* wr = wbase + k0;
    bf16* bt = &Bs[buf][0];
#pragma unroll
    for (int i = 0; i < 8; ++i)
      store_bpair(bt, dseg * 16 + 2 * i, dl, wr[2 * i], wr[2 * i + 1]);
  };

  stage(0, 0);
  async_wait0();
  __syncthreads();
  constexpr int NK = HH / 32;
  for (int it = 0; it < NK; ++it) {
    int cur = it & 1;
    if (it + 1 < NK) stage(cur ^ 1, (it + 1) * 32);
    mma_128x64(As[cur], Bs[cur], lane, wm, acc);
    async_wait0();
    __syncthreads();
  }
  bf16* outb = slot_out + (size_t)b * ESS * DD;
#pragma unroll
  for (int i = 0; i < 2; ++i)
#pragma unroll
    for (int j = 0; j < 4; ++j) {
      int nn = n0 + j * 16 + (lane & 15);
      int mh = m0 + wm * 32 + i * 16 + ((lane >> 4) ? 8 : 0);
      float bias = b2[e * DD + nn];
#pragma unroll
      for (int r = 0; r < 8; ++r)
        outb[(size_t)(mh + r) * DD + nn] = (bf16)(acc[i][j][r] + bias);
    }
}

// ===========================================================================
// GEMM4: out = combine @ slot_out  (N x D, K = ES).
__global__ __launch_bounds__(128) void k_gemm_out(
    const bf16* __restrict__ logits, const bf16* __restrict__ slot_out,
    const float* __restrict__ rowmax, const float* __restrict__ rowinv,
    float* __restrict__ out) {
  __shared__ bf16 As[2][128 * LDA];
  __shared__ bf16 Bs[2][32 * 64];
  int tid = threadIdx.x, lane = tid & 31, wm = tid >> 5;
  int b = blockIdx.z;
  int m0 = blockIdx.y * 128;  // token tile
  int n0 = blockIdx.x * 64;   // d tile
  v8f acc[2][4];
#pragma unroll
  for (int i = 0; i < 2; ++i)
#pragma unroll
    for (int j = 0; j < 4; ++j) acc[i][j] = vzero8();

  int kp = tid >> 3, ng = (tid & 7) * 8;  // B pairs: 2 es-rows x 8 d
  const float rm = rowmax[b * NN + m0 + tid];
  const float ri = rowinv[b * NN + m0 + tid];
  const bf16* lbase = logits + ((size_t)(b * NN + m0 + tid)) * ESS;

  auto stage = [&](int buf, int k0) {
    const uint4* lp = (const uint4*)(lbase + k0);  // 32 bf16 = 4 uint4
    bf16* ar = &As[buf][tid * LDA];
#pragma unroll
    for (int q = 0; q < 4; ++q) {
      uint4 u = lp[q];
      unsigned uu[4] = {u.x, u.y, u.z, u.w};
#pragma unroll
      for (int w = 0; w < 4; ++w) {
        ar[q * 8 + 2 * w]     = (bf16)(__expf(bflo(uu[w]) - rm) * ri);
        ar[q * 8 + 2 * w + 1] = (bf16)(__expf(bfhi(uu[w]) - rm) * ri);
      }
    }
    const bf16* sb0 =
        slot_out + ((size_t)(b * ESS + k0 + 2 * kp)) * DD + n0 + ng;
    const bf16* sb1 = sb0 + DD;
    bf16* bt = &Bs[buf][0];
#pragma unroll
    for (int i = 0; i < 8; ++i)
      store_bpair(bt, 2 * kp, ng + i, sb0[i], sb1[i]);
  };

  stage(0, 0);
  __syncthreads();
  constexpr int NK = ESS / 32;
  for (int it = 0; it < NK; ++it) {
    int cur = it & 1;
    if (it + 1 < NK) stage(cur ^ 1, (it + 1) * 32);
    mma_128x64(As[cur], Bs[cur], lane, wm, acc);
    __syncthreads();
  }
  float* outb = out + (size_t)b * NN * DD;
#pragma unroll
  for (int i = 0; i < 2; ++i)
#pragma unroll
    for (int j = 0; j < 4; ++j)
      store_c_f32(outb, DD, m0 + wm * 32 + i * 16, n0 + j * 16, acc[i][j], lane);
}

// ===========================================================================
extern "C" void kernel_launch(void* const* d_in, const int* in_sizes, int n_in,
                              void* d_out, int out_size, void* d_ws,
                              size_t ws_size, hipStream_t stream) {
  const float* x     = (const float*)d_in[0];
  const float* mu    = (const float*)d_in[1];
  const float* scale = (const float*)d_in[2];
  const float* w1    = (const float*)d_in[3];
  const float* b1    = (const float*)d_in[4];
  const float* w2    = (const float*)d_in[5];
  const float* b2    = (const float*)d_in[6];
  float* out = (float*)d_out;

  size_t off = 0;
  auto take = [&](size_t bytes) {
    char* p = (char*)d_ws + off;
    off += (bytes + 255) & ~(size_t)255;
    return (void*)p;
  };
  bf16*  logits   = (bf16*)take((size_t)BB * NN * ESS * 2);   // 128 MB
  bf16*  slot_in  = (bf16*)take((size_t)BB * ESS * DD * 2);   // 32 MB
  bf16*  hbuf     = (bf16*)take((size_t)BB * ESS * HH * 2);   // 16 MB
  bf16*  slot_out = (bf16*)take((size_t)BB * ESS * DD * 2);   // 32 MB
  bf16*  w1b      = (bf16*)take((size_t)EE * HH * DD * 2);    // 8 MB
  bf16*  w2b      = (bf16*)take((size_t)EE * DD * HH * 2);    // 8 MB
  float* rnx      = (float*)take((size_t)BB * NN * 4);
  float* cmu      = (float*)take((size_t)ESS * 4);
  float* rowmax   = (float*)take((size_t)BB * NN * 4);
  float* rowinv   = (float*)take((size_t)BB * NN * 4);
  float* colmax   = (float*)take((size_t)BB * ESS * 4);
  float* colinv   = (float*)take((size_t)BB * ESS * 4);

  k_rnx<<<(BB * NN) / 8, 256, 0, stream>>>(x, rnx);
  k_cmu<<<ESS / 256, 256, 0, stream>>>(mu, scale, cmu);
  {
    int n1 = EE * HH * DD;
    k_cvt_bf16<<<(n1 + 255) / 256, 256, 0, stream>>>(w1, w1b, n1);
    int n2 = EE * DD * HH;
    k_cvt_bf16<<<(n2 + 255) / 256, 256, 0, stream>>>(w2, w2b, n2);
  }

  k_gemm_logits<<<dim3(ESS / 64, NN / 128, BB), 128, 0, stream>>>(x, mu, rnx,
                                                                  cmu, logits);
  k_rowstats<<<(BB * NN) / 8, 256, 0, stream>>>(logits, rowmax, rowinv);
  k_colstats<<<dim3(ESS / 256, BB), 256, 0, stream>>>(logits, colmax, colinv);

  k_gemm_slot_in<<<dim3(DD / 64, ESS / 128, BB), 128, 0, stream>>>(
      logits, x, colmax, colinv, slot_in);

  k_gemm_mlp1<<<dim3(HH / 64, ESS / 128, BB), 128, 0, stream>>>(slot_in, w1b,
                                                                b1, hbuf);
  k_gemm_mlp2<<<dim3(DD / 64, ESS / 128, BB), 128, 0, stream>>>(hbuf, w2b, b2,
                                                                slot_out);

  k_gemm_out<<<dim3(DD / 64, NN / 128, BB), 128, 0, stream>>>(
      logits, slot_out, rowmax, rowinv, out);
}
